// PointTransformer_75522704933010
// MI455X (gfx1250) — compile-verified, matrix operand order checked by reading, not run on hardware
//
#include <hip/hip_runtime.h>
#include <hip/hip_bf16.h>

// ---------------------------------------------------------------------------
// PointTransformer pipeline for MI455X (gfx1250, wave32).
// Dense projections -> TDM tensor_load_to_lds + v_wmma_f32_16x16x32_f16.
// KNN / softmax-attn / BN / upsample / edge-MLP -> VALU kernels.
// ---------------------------------------------------------------------------

typedef __attribute__((ext_vector_type(16))) _Float16 v16h;
typedef __attribute__((ext_vector_type(8)))  float    v8f;
typedef __attribute__((ext_vector_type(4)))  unsigned int v4u;
typedef __attribute__((ext_vector_type(8)))  int      v8i;
typedef __attribute__((ext_vector_type(4)))  int      v4i;

// Problem constants (match reference setup_inputs: B=4, N=512)
#define BB   4
#define N0   512
#define N1   1536          // 3*N0
#define N2   4608          // 9*N0  (= Nf)
#define NF   4608
#define KNN_K 16
#define E_TOT (BB * NF * KNN_K)     // 294912
#define OUT0  (BB * NF * 3)         // 55296 floats: `output`

// ---------------------------------------------------------------------------
// Tensor Data Mover: 2D tile load Global -> LDS with hardware zero-fill of
// out-of-tensor reads (replaces all software bounds guards).
//   tensor_d0/d1 : valid extents (elements) from the tile origin
//   tile_d0/d1   : tile extents written to LDS (contiguous, row-major)
//   stride_elems : row stride of the source tensor (elements)
// ---------------------------------------------------------------------------
__device__ static inline unsigned int lds_off_of(const void* p)
{
    return (unsigned int)(uintptr_t)p;   // low 32 bits of flat LDS address
}

__device__ static inline void tdm_load_tile(unsigned int lds_off, const float* gsrc,
                                            unsigned int tensor_d0, unsigned int tensor_d1,
                                            unsigned int tile_d0, unsigned int tile_d1,
                                            unsigned int stride_elems)
{
    unsigned long long ga = (unsigned long long)(uintptr_t)gsrc;

    v4u g0;
    g0[0] = 1u;                                   // count=1, user descriptor
    g0[1] = lds_off;                              // lds_addr (bytes)
    g0[2] = (unsigned int)(ga & 0xFFFFFFFFu);     // global_addr[31:0]
    g0[3] = ((unsigned int)(ga >> 32) & 0x01FFFFFFu) | 0x80000000u; // addr[56:32] | type=2

    v8i g1;
    g1[0] = (int)(2u << 16);                                  // data_size = 4B; mask=0
    g1[1] = (int)(tensor_d0 << 16);                           // tensor_dim0[15:0] -> [31:16]
    g1[2] = (int)((tensor_d0 >> 16) | (tensor_d1 << 16));     // dim0[31:16] | dim1[15:0]
    g1[3] = (int)((tensor_d1 >> 16) | (tile_d0 << 16));       // dim1[31:16] | tile_dim0
    g1[4] = (int)(tile_d1 & 0xFFFFu);                         // tile_dim1; tile_dim2=0
    g1[5] = (int)stride_elems;                                // tensor_dim0_stride[31:0]
    g1[6] = 0;                                                // stride[47:32] | dim1_stride lo
    g1[7] = 0;

    v4i z = {0, 0, 0, 0};
#if defined(__clang_major__) && (__clang_major__ >= 23)
    v8i z8 = {0, 0, 0, 0, 0, 0, 0, 0};
    __builtin_amdgcn_tensor_load_to_lds(g0, g1, z, z, z8, 0);
#else
    __builtin_amdgcn_tensor_load_to_lds(g0, g1, z, z, 0);
#endif
}

// ---------------------------------------------------------------------------
// WMMA GEMM:  Y[M,Co] = X[M,Ci] @ W[Ci,Co] + bias   (M multiple of 16, Ci<=64)
// One wave per 16x16 output tile. All K-tiles TDM-loaded up front into
// disjoint LDS regions (no WAR hazard), one tensorcnt wait, peeled WMMAs.
// ---------------------------------------------------------------------------
__device__ static inline void frag_mma(const float* __restrict__ As,
                                       const float* __restrict__ Bs,
                                       v8f& acc, int lane)
{
    const int row16 = lane & 15;
    const int kb    = (lane >> 4) * 8;
    v16h a, b;
    #pragma unroll
    for (int j = 0; j < 8; ++j) {
        #pragma unroll
        for (int h = 0; h < 2; ++h) {
            int kk = (j < 4) ? (kb + 2 * j + h) : (kb + 16 + 2 * (j - 4) + h);
            a[2 * j + h] = (_Float16)As[row16 * 32 + kk];
            b[2 * j + h] = (_Float16)Bs[kk * 16 + row16];
        }
    }
    acc = __builtin_amdgcn_wmma_f32_16x16x32_f16(
        false, a, false, b, (short)0, acc, false, false);
}

__global__ __launch_bounds__(32)
void gemm_bias_kernel(const float* __restrict__ X, const float* __restrict__ W,
                      const float* __restrict__ bias, float* __restrict__ Y,
                      int M, int Ci, int Co)
{
    __shared__ float As[2][16 * 32];   // two K-tiles of A (16 rows x 32 K)
    __shared__ float Bs[2][32 * 16];   // two K-tiles of B (32 K x 16 cols)

    const int lane   = threadIdx.x;          // 0..31
    const int tile_m = blockIdx.x;           // M/16 tiles
    const int tile_n = blockIdx.y;           // ceil(Co/16) tiles
    const int row0   = tile_m * 16;
    const int col0   = tile_n * 16;
    const int ktiles = (Ci + 31) >> 5;       // 1 or 2

    // Async TDM loads: hardware zero-fills beyond (Ci - k0) / (Co - col0).
    tdm_load_tile(lds_off_of(As[0]), X + (size_t)row0 * Ci,
                  (unsigned)Ci, (unsigned)(M - row0), 32u, 16u, (unsigned)Ci);
    tdm_load_tile(lds_off_of(Bs[0]), W + col0,
                  (unsigned)(Co - col0), (unsigned)Ci, 16u, 32u, (unsigned)Co);
    if (ktiles > 1) {
        tdm_load_tile(lds_off_of(As[1]), X + (size_t)row0 * Ci + 32,
                      (unsigned)(Ci - 32), (unsigned)(M - row0), 32u, 16u, (unsigned)Ci);
        tdm_load_tile(lds_off_of(Bs[1]), W + (size_t)32 * Co + col0,
                      (unsigned)(Co - col0), (unsigned)(Ci - 32), 16u, 32u, (unsigned)Co);
    }
    __builtin_amdgcn_s_wait_tensorcnt(0);
    asm volatile("" ::: "memory");           // LDS now holds TDM-written tiles
    __syncthreads();

    v8f acc = {0.f, 0.f, 0.f, 0.f, 0.f, 0.f, 0.f, 0.f};
    frag_mma(As[0], Bs[0], acc, lane);
    if (ktiles > 1) frag_mma(As[1], Bs[1], acc, lane);

    const int col = col0 + (lane & 15);
    if (col < Co) {
        const float bv = bias ? bias[col] : 0.0f;
        const int rofs = (lane >> 4) * 8;
        #pragma unroll
        for (int r = 0; r < 8; ++r) {
            int m = row0 + r + rofs;
            Y[m * Co + col] = acc[r] + bv;
        }
    }
}

// ---------------------------------------------------------------------------
// KNN: per point, scan its batch, keep 16 smallest squared distances
// (self excluded). Writes GLOBAL row indices.
// ---------------------------------------------------------------------------
template <int C>
__global__ void knn_kernel(const float* __restrict__ x, int Np, int total,
                           int* __restrict__ idx_out)
{
    int t = blockIdx.x * blockDim.x + threadIdx.x;
    if (t >= total) return;
    const int n    = t % Np;
    const int base = t - n;            // batch start row

    float qv[C];
    #pragma unroll
    for (int i = 0; i < C; ++i) qv[i] = x[t * C + i];

    float bd[KNN_K];
    int   bi[KNN_K];
    #pragma unroll
    for (int j = 0; j < KNN_K; ++j) { bd[j] = 3.0e38f; bi[j] = base; }

    for (int m = 0; m < Np; ++m) {
        if (m + 8 < Np) __builtin_prefetch(&x[(size_t)(base + m + 8) * C], 0, 1);
        if (m == n) continue;
        const float* xm = &x[(size_t)(base + m) * C];
        float d = 0.0f;
        #pragma unroll
        for (int i = 0; i < C; ++i) { float df = qv[i] - xm[i]; d += df * df; }
        if (d < bd[KNN_K - 1]) {
            bd[KNN_K - 1] = d; bi[KNN_K - 1] = base + m;
            #pragma unroll
            for (int j = KNN_K - 1; j >= 1; --j) {
                if (bd[j] < bd[j - 1]) {
                    float td = bd[j]; bd[j] = bd[j - 1]; bd[j - 1] = td;
                    int   ti = bi[j]; bi[j] = bi[j - 1]; bi[j - 1] = ti;
                }
            }
        }
    }
    #pragma unroll
    for (int j = 0; j < KNN_K; ++j) idx_out[t * KNN_K + j] = bi[j];
}

// ---------------------------------------------------------------------------
// Vector attention: out[m,c] = sum_j softmax_j(q[m,c]-k[idx[m,j],c]) * v[...]
// ---------------------------------------------------------------------------
__global__ void attn_kernel(const float* __restrict__ Q, const float* __restrict__ Kf,
                            const float* __restrict__ Vf, const int* __restrict__ idx,
                            float* __restrict__ out, int M, int C)
{
    int t = blockIdx.x * blockDim.x + threadIdx.x;
    if (t >= M * C) return;
    const int m = t / C, c = t % C;
    const float q = Q[m * C + c];

    float s[KNN_K], vv[KNN_K];
    float mx = -3.0e38f;
    #pragma unroll
    for (int j = 0; j < KNN_K; ++j) {
        int g = idx[m * KNN_K + j];
        float sv = q - Kf[g * C + c];
        s[j]  = sv;
        vv[j] = Vf[g * C + c];
        mx = fmaxf(mx, sv);
    }
    float den = 0.0f, num = 0.0f;
    #pragma unroll
    for (int j = 0; j < KNN_K; ++j) {
        float e = __expf(s[j] - mx);
        den += e;
        num += e * vv[j];
    }
    out[m * C + c] = num / den;
}

// ---------------------------------------------------------------------------
// BatchNorm (batch stats, biased var) + LeakyReLU(0.2), inplace. Block/channel.
// ---------------------------------------------------------------------------
__global__ __launch_bounds__(256)
void bn_lrelu_kernel(float* __restrict__ Y, const float* __restrict__ g,
                     const float* __restrict__ b, int M, int C)
{
    const int c   = blockIdx.x;
    const int tid = threadIdx.x;
    __shared__ float r1[256], r2[256];

    float s = 0.0f, s2 = 0.0f;
    for (int i = tid; i < M; i += 256) {
        float v = Y[(size_t)i * C + c];
        s += v; s2 += v * v;
    }
    r1[tid] = s; r2[tid] = s2;
    __syncthreads();
    for (int off = 128; off > 0; off >>= 1) {
        if (tid < off) { r1[tid] += r1[tid + off]; r2[tid] += r2[tid + off]; }
        __syncthreads();
    }
    const float mean = r1[0] / (float)M;
    const float var  = r2[0] / (float)M - mean * mean;
    const float inv  = 1.0f / sqrtf(var + 1e-5f);
    const float gg = g[c], bb = b[c];

    for (int i = tid; i < M; i += 256) {
        float y = (Y[(size_t)i * C + c] - mean) * inv * gg + bb;
        Y[(size_t)i * C + c] = (y >= 0.0f) ? y : 0.2f * y;
    }
}

// ---------------------------------------------------------------------------
// Linear upsample x3 along point axis, align_corners=True.
// ---------------------------------------------------------------------------
__global__ void upsample3_kernel(const float* __restrict__ in, float* __restrict__ out,
                                 int Bb, int L, int C)
{
    const int Lo = 3 * L;
    int t = blockIdx.x * blockDim.x + threadIdx.x;
    if (t >= Bb * Lo * C) return;
    const int c   = t % C;
    const int o   = (t / C) % Lo;
    const int bch = t / (Lo * C);
    float pos = (float)o * (float)(L - 1) / (float)(Lo - 1);
    int lo = (int)floorf(pos);
    int hi = lo + 1; if (hi > L - 1) hi = L - 1;
    float w = pos - (float)lo;
    out[t] = in[((size_t)bch * L + lo) * C + c] * (1.0f - w)
           + in[((size_t)bch * L + hi) * C + c] * w;
}

// ---------------------------------------------------------------------------
// Edge stage: edge_index + fused edge MLP  (6 -> 32 -> 1, LeakyReLU(0.2)).
// ---------------------------------------------------------------------------
__global__ void edge_kernel(const float* __restrict__ edge_in, const int* __restrict__ idx,
                            const float* __restrict__ w1, const float* __restrict__ b1,
                            const float* __restrict__ w2, const float* __restrict__ b2,
                            int* __restrict__ ei, float* __restrict__ eout, int E)
{
    int t = blockIdx.x * blockDim.x + threadIdx.x;
    if (t >= E) return;
    const int dst = t / KNN_K;      // global row (b*Nf + n)
    const int src = idx[t];         // already global
    ei[t]     = src;
    ei[E + t] = dst;

    float in6[6];
    #pragma unroll
    for (int i = 0; i < 3; ++i) in6[i]     = edge_in[src * 3 + i];
    #pragma unroll
    for (int i = 0; i < 3; ++i) in6[3 + i] = edge_in[dst * 3 + i];

    float o = b2[0];
    #pragma unroll
    for (int h = 0; h < 32; ++h) {
        float a = b1[h];
        #pragma unroll
        for (int i = 0; i < 6; ++i) a += in6[i] * w1[i * 32 + h];
        a = (a >= 0.0f) ? a : 0.2f * a;
        o += a * w2[h];
    }
    eout[t] = o;
}

// ---------------------------------------------------------------------------
// Host-side orchestration (all launches on `stream`; no allocation).
// ---------------------------------------------------------------------------
static inline void launch_gemm(const float* X, const float* W, const float* b,
                               float* Y, int M, int Ci, int Co, hipStream_t s)
{
    dim3 grid(M / 16, (Co + 15) / 16);
    gemm_bias_kernel<<<grid, 32, 0, s>>>(X, W, b, Y, M, Ci, Co);
}

static inline void launch_knn(const float* x, int Np, int total, int C,
                              int* idx, hipStream_t s)
{
    int blocks = (total + 127) / 128;
    if (C == 64)      knn_kernel<64><<<blocks, 128, 0, s>>>(x, Np, total, idx);
    else if (C == 32) knn_kernel<32><<<blocks, 128, 0, s>>>(x, Np, total, idx);
    else              knn_kernel<3 ><<<blocks, 128, 0, s>>>(x, Np, total, idx);
}

static inline void run_pt_layer(const float* xin, int Np, int C,
                                const float* wq, const float* bq,
                                const float* wk, const float* bk,
                                const float* wv, const float* bv,
                                float* Qb, float* Kb, float* Vb, int* IDX,
                                bool doKnn, float* out, hipStream_t s)
{
    const int M = BB * Np;
    if (doKnn) launch_knn(xin, Np, M, C, IDX, s);
    launch_gemm(xin, wq, bq, Qb, M, C, C, s);
    launch_gemm(xin, wk, bk, Kb, M, C, C, s);
    launch_gemm(xin, wv, bv, Vb, M, C, C, s);
    int total = M * C;
    attn_kernel<<<(total + 255) / 256, 256, 0, s>>>(Qb, Kb, Vb, IDX, out, M, C);
}

extern "C" void kernel_launch(void* const* d_in, const int* in_sizes, int n_in,
                              void* d_out, int out_size, void* d_ws, size_t ws_size,
                              hipStream_t stream)
{
    (void)in_sizes; (void)n_in; (void)out_size; (void)ws_size;

    const float* features = (const float*)d_in[0];   // [4,512,64]
    // d_in[1] = points (unused by the reference forward pass)
    auto P = [&](int i) { return (const float*)d_in[i]; };

    // param indices (flattened in setup_inputs insertion order)
    const int B11 = 2, B12 = 8, B21 = 14, B22 = 20, B31 = 26, B32 = 32, EDG = 38;
    const float* bn64_g = P(44); const float* bn64_b = P(45);
    const float* bn32_g = P(46); const float* bn32_b = P(47);
    const float* lin1_w = P(48); const float* lin1_b = P(49);
    const float* l1bn_g = P(50); const float* l1bn_b = P(51);
    const float* lin2_w = P(52); const float* lin2_b = P(53);
    const float* l2bn_g = P(54); const float* l2bn_b = P(55);
    const float* ew1 = P(56); const float* eb1 = P(57);
    const float* ew2 = P(58); const float* eb2 = P(59);

    // workspace layout (floats)
    const size_t SLOT = (size_t)BB * NF * 64;   // 1,179,648 floats
    float* ws = (float*)d_ws;
    float* X  = ws;
    float* Y  = ws + SLOT;
    float* Qb = ws + 2 * SLOT;
    float* Kb = ws + 3 * SLOT;
    float* Vb = ws + 4 * SLOT;
    int*   IDX = (int*)(ws + 5 * SLOT);                       // [B*Nf, 16]
    float* edge_in = ws + 5 * SLOT + (size_t)BB * NF * KNN_K; // [B*Nf, 3]

    float* outF = (float*)d_out;
    float* output = outF;                          // [B, Nf, 3]
    int*   ei     = (int*)(outF + OUT0);           // [2, E] (int bits)
    float* eout   = outF + OUT0 + 2 * E_TOT;       // [E, 1]

    // ---- block1: two pt-layers at N=512, C=64 + BN64 ----
    run_pt_layer(features, N0, 64, P(B11), P(B11+1), P(B11+2), P(B11+3), P(B11+4), P(B11+5),
                 Qb, Kb, Vb, IDX, true, Y, stream);
    bn_lrelu_kernel<<<64, 256, 0, stream>>>(Y, bn64_g, bn64_b, BB * N0, 64);
    run_pt_layer(Y, N0, 64, P(B12), P(B12+1), P(B12+2), P(B12+3), P(B12+4), P(B12+5),
                 Qb, Kb, Vb, IDX, true, X, stream);
    bn_lrelu_kernel<<<64, 256, 0, stream>>>(X, bn64_g, bn64_b, BB * N0, 64);

    // ---- upsample x3: [4,512,64] -> [4,1536,64] ----
    {
        int total = BB * N1 * 64;
        upsample3_kernel<<<(total + 255) / 256, 256, 0, stream>>>(X, Y, BB, N0, 64);
    }

    // ---- block2_1 at N=1536, C=64 + BN64 ----
    run_pt_layer(Y, N1, 64, P(B21), P(B21+1), P(B21+2), P(B21+3), P(B21+4), P(B21+5),
                 Qb, Kb, Vb, IDX, true, X, stream);
    bn_lrelu_kernel<<<64, 256, 0, stream>>>(X, bn64_g, bn64_b, BB * N1, 64);

    // ---- lin1: 64 -> 32 + BN ----
    launch_gemm(X, lin1_w, lin1_b, Y, BB * N1, 64, 32, stream);
    bn_lrelu_kernel<<<32, 256, 0, stream>>>(Y, l1bn_g, l1bn_b, BB * N1, 32);

    // ---- block2_2 at N=1536, C=32 + BN32 ----
    run_pt_layer(Y, N1, 32, P(B22), P(B22+1), P(B22+2), P(B22+3), P(B22+4), P(B22+5),
                 Qb, Kb, Vb, IDX, true, X, stream);
    bn_lrelu_kernel<<<32, 256, 0, stream>>>(X, bn32_g, bn32_b, BB * N1, 32);

    // ---- upsample x3: [4,1536,32] -> [4,4608,32] ----
    {
        int total = BB * N2 * 32;
        upsample3_kernel<<<(total + 255) / 256, 256, 0, stream>>>(X, Y, BB, N1, 32);
    }

    // ---- block3_1 at N=4608, C=32 + BN32 ----
    run_pt_layer(Y, N2, 32, P(B31), P(B31+1), P(B31+2), P(B31+3), P(B31+4), P(B31+5),
                 Qb, Kb, Vb, IDX, true, X, stream);
    bn_lrelu_kernel<<<32, 256, 0, stream>>>(X, bn32_g, bn32_b, BB * N2, 32);

    // ---- lin2: 32 -> 3 + BN ----
    launch_gemm(X, lin2_w, lin2_b, Y, BB * N2, 32, 3, stream);
    bn_lrelu_kernel<<<3, 256, 0, stream>>>(Y, l2bn_g, l2bn_b, BB * N2, 3);

    // ---- block3_2 at N=4608, C=3 -> output (no BN) ----
    run_pt_layer(Y, N2, 3, P(B32), P(B32+1), P(B32+2), P(B32+3), P(B32+4), P(B32+5),
                 Qb, Kb, Vb, IDX, true, output, stream);

    // ---- KNN on `output` once: reused by edge_block pt-layer AND edge_index ----
    launch_knn(output, N2, BB * N2, 3, IDX, stream);

    // ---- edge_block pt-layer (C=3) on output, KNN precomputed ----
    run_pt_layer(output, N2, 3, P(EDG), P(EDG+1), P(EDG+2), P(EDG+3), P(EDG+4), P(EDG+5),
                 Qb, Kb, Vb, IDX, false, edge_in, stream);

    // ---- edge_index + fused edge MLP ----
    edge_kernel<<<(E_TOT + 255) / 256, 256, 0, stream>>>(
        edge_in, IDX, ew1, eb1, ew2, eb2, ei, eout, E_TOT);
}
